// GraphAttentionEncoder_8409545966421
// MI455X (gfx1250) — matrix operand order repaired and checked by reading, not compiled, not run on hardware
//
#include <hip/hip_runtime.h>
#include <hip/hip_bf16.h>

// GraphAttentionEncoder for gfx1250 (MI455X): bf16 WMMA GEMMs + wave32 LN +
// atomic segment-softmax edge phase.

#define HGAT_H     8
#define HGAT_C     32
#define HGAT_D     256
#define HGAT_INDIM 128

typedef __attribute__((ext_vector_type(16))) __bf16 v16bf;
typedef __attribute__((ext_vector_type(8)))  __bf16 v8bf;
typedef __attribute__((ext_vector_type(8)))  float  v8f;

__device__ __forceinline__ unsigned short f32_to_bf16(float f) {
  unsigned int u = __float_as_uint(f);
  u += 0x7fffu + ((u >> 16) & 1u);   // round-to-nearest-even
  return (unsigned short)(u >> 16);
}

__device__ __forceinline__ float wave_sum(float v) {
#pragma unroll
  for (int off = 16; off > 0; off >>= 1) v += __shfl_xor(v, off, 32);
  return v;
}

__device__ __forceinline__ void atomic_max_f32(float* addr, float val) {
  unsigned int* ua = (unsigned int*)addr;
  unsigned int old = *ua;
  while (__uint_as_float(old) < val) {
    unsigned int assumed = old;
    old = atomicCAS(ua, assumed, __float_as_uint(val));
    if (old == assumed) break;
  }
}

// ---------------------------------------------------------------- conversion
__global__ void k_f32_to_bf16(const float* __restrict__ in,
                              unsigned short* __restrict__ out, int n) {
  int i = blockIdx.x * blockDim.x + threadIdx.x;
  if (i < n) out[i] = f32_to_bf16(in[i]);
}

// Wt[c, r] = bf16(W[r, c]) : makes B-fragment loads contiguous in K.
__global__ void k_transpose_bf16(const float* __restrict__ W,
                                 unsigned short* __restrict__ Wt,
                                 int rows, int cols) {
  int i = blockIdx.x * blockDim.x + threadIdx.x;
  if (i >= rows * cols) return;
  int r = i / cols, c = i % cols;
  Wt[(size_t)c * rows + r] = f32_to_bf16(W[i]);
}

// ---------------------------------------------------------------- WMMA GEMM
// C[M,Nn] = A[M,K] (bf16, row-major) * B[K,Nn], B given transposed as Bt[Nn,K].
// One wave -> 16x64 strip (4 accumulators, A fragment reused 4x).
// Block = 128 thr = 4 waves -> 16 x 256 per block. grid = (M/16, Nn/256).
__global__ __launch_bounds__(128)
void k_gemm_wmma_bf16(const unsigned short* __restrict__ A,
                      const unsigned short* __restrict__ Bt,
                      float* __restrict__ Cd, int M, int Nn, int K) {
  const int lane  = threadIdx.x & 31;
  const int wave  = threadIdx.x >> 5;
  const int mBase = blockIdx.x * 16;
  const int nBase = blockIdx.y * 256 + wave * 64;
  if (nBase >= Nn) return;

  const int mRow  = lane & 15;          // A: M across lanes 0..15
  const int kOffA = (lane >> 4) * 16;   // A: lanes 0-15 K=0..15, 16-31 K=16..31
  const int kOffB = (lane >> 4) * 16;   // B: lanes 0-15 K=0..15, 16-31 K=16..31
  const int nCol  = lane & 15;          // B/C: N across lanes 0..15

  int arow = mBase + mRow;
  if (arow >= M) arow = M - 1;          // clamp (store is guarded)
  const __bf16* aBase = (const __bf16*)A + (size_t)arow * K;

  v8f acc[4] = {v8f{}, v8f{}, v8f{}, v8f{}};

  for (int k = 0; k < K; k += 32) {
    // A fragment: 16x32 bf16, two contiguous 16B loads per lane.
    v8bf alo = *(const v8bf*)(aBase + k + kOffA);
    v8bf ahi = *(const v8bf*)(aBase + k + kOffA + 8);
    v16bf af;
#pragma unroll
    for (int j = 0; j < 8; ++j) { af[j] = alo[j]; af[j + 8] = ahi[j]; }

#pragma unroll
    for (int t = 0; t < 4; ++t) {
      // B fragment: 32x16 bf16, one contiguous 32B load per lane from Bt.
      const __bf16* bp =
          (const __bf16*)Bt + (size_t)(nBase + t * 16 + nCol) * K + k + kOffB;
      v16bf bfrag = *(const v16bf*)bp;
      acc[t] = __builtin_amdgcn_wmma_f32_16x16x32_bf16(
          false, af, false, bfrag, (short)0, acc[t], false, false);
    }
  }

  // C/D layout: VGPR r, lanes 0-15 -> M=r, lanes 16-31 -> M=r+8; N=lane%16.
  const int rowHalf = (lane >> 4) * 8;
#pragma unroll
  for (int t = 0; t < 4; ++t)
#pragma unroll
    for (int r = 0; r < 8; ++r) {
      int row = mBase + rowHalf + r;
      if (row < M) Cd[(size_t)row * Nn + nBase + t * 16 + nCol] = acc[t][r];
    }
}

// ---------------------------------------------------------------- attention
__global__ void k_attn_logits(const float* __restrict__ hl,
                              const float* __restrict__ a_src,
                              const float* __restrict__ a_dst,
                              float* __restrict__ al_s,
                              float* __restrict__ al_d, int N) {
  int i = blockIdx.x * blockDim.x + threadIdx.x;
  if (i >= N * HGAT_H) return;
  int n = i / HGAT_H, h = i % HGAT_H;
  const float* row = hl + (size_t)n * HGAT_D + h * HGAT_C;
  const float* as  = a_src + h * HGAT_C;
  const float* ad  = a_dst + h * HGAT_C;
  float ss = 0.f, sd = 0.f;
#pragma unroll
  for (int c = 0; c < HGAT_C; ++c) {
    float v = row[c];
    ss += v * as[c];
    sd += v * ad[c];
  }
  al_s[i] = ss;
  al_d[i] = sd;
}

__global__ void k_init_layer(float* __restrict__ m, float* __restrict__ denom,
                             float* __restrict__ out_acc, int N) {
  int i = blockIdx.x * blockDim.x + threadIdx.x;
  if (i < N * HGAT_D) out_acc[i] = 0.f;
  if (i < N * HGAT_H) { m[i] = -1e30f; denom[i] = 0.f; }
}

// Pass 1: e = leaky_relu(al_s[s]+al_d[d]); segment max over dst.
__global__ void k_edge_logit_max(const int* __restrict__ src,
                                 const int* __restrict__ dst,
                                 const float* __restrict__ al_s,
                                 const float* __restrict__ al_d,
                                 float* __restrict__ e_buf,
                                 float* __restrict__ m, int E, int E2) {
  int i = blockIdx.x * blockDim.x + threadIdx.x;
  if (i >= E2) return;
  int s = (i < E) ? src[i] : (i - E);   // self loops appended
  int d = (i < E) ? dst[i] : (i - E);
#pragma unroll
  for (int h = 0; h < HGAT_H; ++h) {
    float e = al_s[s * HGAT_H + h] + al_d[d * HGAT_H + h];
    e = (e > 0.f) ? e : 0.2f * e;
    e_buf[(size_t)i * HGAT_H + h] = e;
    atomic_max_f32(&m[d * HGAT_H + h], e);
  }
}

// Pass 2: ex = exp(e - m[d]); segment sum over dst.
__global__ void k_edge_exp_sum(const int* __restrict__ dst,
                               float* __restrict__ e_buf,
                               const float* __restrict__ m,
                               float* __restrict__ denom, int E, int E2) {
  int i = blockIdx.x * blockDim.x + threadIdx.x;
  if (i >= E2) return;
  int d = (i < E) ? dst[i] : (i - E);
#pragma unroll
  for (int h = 0; h < HGAT_H; ++h) {
    float ex = expf(e_buf[(size_t)i * HGAT_H + h] - m[d * HGAT_H + h]);
    e_buf[(size_t)i * HGAT_H + h] = ex;
    atomicAdd(&denom[d * HGAT_H + h], ex);
  }
}

// Pass 3: out[d] += alpha * hl[s]   (one thread per edge x head, 32 channels)
__global__ void k_edge_scatter(const int* __restrict__ src,
                               const int* __restrict__ dst,
                               const float* __restrict__ e_buf,
                               const float* __restrict__ denom,
                               const float* __restrict__ hl,
                               float* __restrict__ out_acc, int E, int E2) {
  int i = blockIdx.x * blockDim.x + threadIdx.x;
  if (i >= E2 * HGAT_H) return;
  int eIdx = i / HGAT_H, h = i % HGAT_H;
  int s = (eIdx < E) ? src[eIdx] : (eIdx - E);
  int d = (eIdx < E) ? dst[eIdx] : (eIdx - E);
  float alpha = e_buf[i] / (denom[d * HGAT_H + h] + 1e-16f);
  const float* hrow = hl + (size_t)s * HGAT_D + h * HGAT_C;
  float* orow = out_acc + (size_t)d * HGAT_D + h * HGAT_C;
#pragma unroll
  for (int c = 0; c < HGAT_C; ++c) atomicAdd(&orow[c], alpha * hrow[c]);
}

// ------------------------------------------------- LayerNorm / activations
// One wave per node over D=256 (8 elems per lane). act: 0 none, 1 relu, 2 elu.
__global__ __launch_bounds__(256)
void k_ln_post(const float* __restrict__ inp, const float* __restrict__ bias,
               const float* __restrict__ g, const float* __restrict__ b,
               const float* __restrict__ residual, float* __restrict__ outp,
               int N, int act) {
  int wid  = (blockIdx.x * blockDim.x + threadIdx.x) >> 5;
  int lane = threadIdx.x & 31;
  if (wid >= N) return;
  const float* row = inp + (size_t)wid * HGAT_D;
  float v[8];
  float s = 0.f;
#pragma unroll
  for (int j = 0; j < 8; ++j) {
    int idx = j * 32 + lane;
    v[j] = row[idx] + bias[idx];
    s += v[j];
  }
  float mu = wave_sum(s) * (1.f / HGAT_D);
  float var = 0.f;
#pragma unroll
  for (int j = 0; j < 8; ++j) { float dd = v[j] - mu; var += dd * dd; }
  var = wave_sum(var) * (1.f / HGAT_D);
  float rs = rsqrtf(var + 1e-5f);
#pragma unroll
  for (int j = 0; j < 8; ++j) {
    int idx = j * 32 + lane;
    float o = (v[j] - mu) * rs * g[idx] + b[idx];
    if (residual) o += residual[(size_t)wid * HGAT_D + idx];
    if (act == 1) o = fmaxf(o, 0.f);
    else if (act == 2) o = (o > 0.f) ? o : (expf(o) - 1.f);
    outp[(size_t)wid * HGAT_D + idx] = o;
  }
}

// Last layer: mean over heads -> [N,32], +bias, LN over 32 (1 elem/lane).
__global__ __launch_bounds__(256)
void k_post_last(const float* __restrict__ out_acc,
                 const float* __restrict__ bias, const float* __restrict__ g,
                 const float* __restrict__ b, float* __restrict__ outp, int N) {
  int wid  = (blockIdx.x * blockDim.x + threadIdx.x) >> 5;
  int lane = threadIdx.x & 31;
  if (wid >= N) return;
  const float* row = out_acc + (size_t)wid * HGAT_D;
  float s = 0.f;
#pragma unroll
  for (int h = 0; h < HGAT_H; ++h) s += row[h * HGAT_C + lane];
  float v = s * (1.f / HGAT_H) + bias[lane];
  float mu = wave_sum(v) * (1.f / 32.f);
  float dd = v - mu;
  float var = wave_sum(dd * dd) * (1.f / 32.f);
  float rs = rsqrtf(var + 1e-5f);
  outp[(size_t)wid * HGAT_C + lane] = dd * rs * g[lane] + b[lane];
}

// ---------------------------------------------------------------- launcher
extern "C" void kernel_launch(void* const* d_in, const int* in_sizes, int n_in,
                              void* d_out, int out_size, void* d_ws,
                              size_t ws_size, hipStream_t stream) {
  (void)n_in; (void)out_size; (void)ws_size;

  const float* x         = (const float*)d_in[0];
  const int*   edges     = (const int*)d_in[1];
  const float* W_in      = (const float*)d_in[2];
  const float* b_in      = (const float*)d_in[3];
  const float* ln_in_g   = (const float*)d_in[4];
  const float* ln_in_b   = (const float*)d_in[5];
  const float* W_gat     = (const float*)d_in[6];
  const float* att_src   = (const float*)d_in[7];
  const float* att_dst   = (const float*)d_in[8];
  const float* bias_cat  = (const float*)d_in[9];
  const float* bias_last = (const float*)d_in[10];
  const float* ln_g_cat  = (const float*)d_in[11];
  const float* ln_b_cat  = (const float*)d_in[12];
  const float* ln_g_last = (const float*)d_in[13];
  const float* ln_b_last = (const float*)d_in[14];

  const int N  = in_sizes[0] / HGAT_INDIM;
  const int E  = in_sizes[1] / 2;
  const int E2 = E + N;
  const int Nd = N * HGAT_D;

  const int* srcP = edges;
  const int* dstP = edges + E;

  // ---- workspace carve (256B aligned slices) ----
  char* base = (char*)d_ws;
  size_t off = 0;
  auto carve = [&](size_t bytes) -> char* {
    char* p = base + off;
    off = (off + bytes + 255) & ~(size_t)255;
    return p;
  };
  float*          hA      = (float*)carve((size_t)Nd * 4);
  float*          hB      = (float*)carve((size_t)Nd * 4);
  float*          hl      = (float*)carve((size_t)Nd * 4);
  float*          outAcc  = (float*)carve((size_t)Nd * 4);
  unsigned short* hb      = (unsigned short*)carve((size_t)Nd * 2);
  unsigned short* wt      = (unsigned short*)carve((size_t)HGAT_D * HGAT_D * 2);
  float*          al_s    = (float*)carve((size_t)N * HGAT_H * 4);
  float*          al_d    = (float*)carve((size_t)N * HGAT_H * 4);
  float*          mBuf    = (float*)carve((size_t)N * HGAT_H * 4);
  float*          denomB  = (float*)carve((size_t)N * HGAT_H * 4);
  float*          e_buf   = (float*)carve((size_t)E2 * HGAT_H * 4);

  auto blocks1d = [](int n) { return dim3((unsigned)((n + 255) / 256)); };
  auto blocksWave = [](int n) { return dim3((unsigned)((n * 32 + 255) / 256)); };

  // ---- input projection: x @ W_in + b_in -> LN -> ReLU ----
  k_f32_to_bf16<<<blocks1d(N * HGAT_INDIM), 256, 0, stream>>>(x, hb, N * HGAT_INDIM);
  k_transpose_bf16<<<blocks1d(HGAT_INDIM * HGAT_D), 256, 0, stream>>>(
      W_in, wt, HGAT_INDIM, HGAT_D);
  {
    dim3 gg((unsigned)((N + 15) / 16), 1);
    k_gemm_wmma_bf16<<<gg, 128, 0, stream>>>(hb, wt, hl, N, HGAT_D, HGAT_INDIM);
  }
  k_ln_post<<<blocksWave(N), 256, 0, stream>>>(hl, b_in, ln_in_g, ln_in_b,
                                               nullptr, hA, N, /*relu*/ 1);

  float* hcur = hA;
  float* hnext = hB;

  for (int i = 0; i < 4; ++i) {
    // h @ W_gat[i]
    k_f32_to_bf16<<<blocks1d(Nd), 256, 0, stream>>>(hcur, hb, Nd);
    k_transpose_bf16<<<blocks1d(HGAT_D * HGAT_D), 256, 0, stream>>>(
        W_gat + (size_t)i * HGAT_D * HGAT_D, wt, HGAT_D, HGAT_D);
    {
      dim3 gg((unsigned)((N + 15) / 16), 1);
      k_gemm_wmma_bf16<<<gg, 128, 0, stream>>>(hb, wt, hl, N, HGAT_D, HGAT_D);
    }

    // attention logits and segment softmax
    k_attn_logits<<<blocks1d(N * HGAT_H), 256, 0, stream>>>(
        hl, att_src + (size_t)i * HGAT_D, att_dst + (size_t)i * HGAT_D,
        al_s, al_d, N);
    k_init_layer<<<blocks1d(Nd), 256, 0, stream>>>(mBuf, denomB, outAcc, N);
    k_edge_logit_max<<<blocks1d(E2), 256, 0, stream>>>(srcP, dstP, al_s, al_d,
                                                       e_buf, mBuf, E, E2);
    k_edge_exp_sum<<<blocks1d(E2), 256, 0, stream>>>(dstP, e_buf, mBuf, denomB,
                                                     E, E2);
    k_edge_scatter<<<blocks1d(E2 * HGAT_H), 256, 0, stream>>>(
        srcP, dstP, e_buf, denomB, hl, outAcc, E, E2);

    if (i < 3) {
      k_ln_post<<<blocksWave(N), 256, 0, stream>>>(
          outAcc, bias_cat + (size_t)i * HGAT_D, ln_g_cat + (size_t)i * HGAT_D,
          ln_b_cat + (size_t)i * HGAT_D, (i > 0) ? hcur : nullptr, hnext, N,
          /*elu*/ 2);
      float* t = hcur; hcur = hnext; hnext = t;
    } else {
      k_post_last<<<blocksWave(N), 256, 0, stream>>>(
          outAcc, bias_last, ln_g_last, ln_b_last, (float*)d_out, N);
    }
  }
}
